// SparseKMoE_29592324669845
// MI455X (gfx1250) — matrix operand order repaired and checked by reading, not compiled, not run on hardware
//
#include <hip/hip_runtime.h>

// ---------------- problem constants ----------------
#define Bb 4
#define Tt 2048
#define Dd 1024
#define Hh 2048
#define Ee 8
#define Nn (Bb*Tt)          // 8192 tokens
#define NA (2*Nn)           // 16384 assignments (top-2)
#define BM 128
#define BN 128
#define BK 32
#define ROWS_CAP (NA + Ee*BM)     // 17408 (per-expert padding to BM)
#define MAXTILES (ROWS_CAP/BM)    // 136

typedef __bf16 bf16;
typedef __bf16 v8bf  __attribute__((ext_vector_type(8)));
typedef __bf16 v16bf __attribute__((ext_vector_type(16)));
typedef float  v8f   __attribute__((ext_vector_type(8)));
typedef float  v4f   __attribute__((ext_vector_type(4)));

__device__ __forceinline__ bf16 f2bf(float f) {
    union { float f; unsigned u; } v; v.f = f;
    unsigned r = v.u + 0x7fffu + ((v.u >> 16) & 1u);   // round-to-nearest-even
    unsigned short h = (unsigned short)(r >> 16);
    return __builtin_bit_cast(bf16, h);
}

// ---- CDNA5 async global->LDS copy (ASYNCcnt path), 16B per lane ----
// the instruction offset applies to both the LDS destination and the global source address
__device__ __forceinline__ void async_cp16(unsigned lds_off, unsigned long long gaddr) {
    asm volatile("global_load_async_to_lds_b128 %0, %1, off"
                 :: "v"(lds_off), "v"(gaddr) : "memory");
}
__device__ __forceinline__ void async_cp16_o16(unsigned lds_off, unsigned long long gaddr) {
    asm volatile("global_load_async_to_lds_b128 %0, %1, off offset:16"
                 :: "v"(lds_off), "v"(gaddr) : "memory");
}
__device__ __forceinline__ void wait_async0() {
    asm volatile("s_wait_asynccnt 0x0" ::: "memory");
}
__device__ __forceinline__ unsigned lds_offset_of(const void* p) {
    // generic LDS address: low 32 bits are the wave-relative LDS byte offset
    return (unsigned)(unsigned long long)(uintptr_t)p;
}

// ---------------- 1) gating: softmax over E, top-2, renormalize ----------------
__global__ void gate_topk_kernel(const float* __restrict__ x,
                                 const float* __restrict__ gate,
                                 int* __restrict__ top2_e,
                                 float* __restrict__ top2_w,
                                 int* __restrict__ cnt) {
    __shared__ float red[Ee * 256];
    const int t   = blockIdx.x;       // token
    const int tid = threadIdx.x;      // 0..255
    float acc[Ee];
#pragma unroll
    for (int e = 0; e < Ee; ++e) acc[e] = 0.f;
    const float* xr = x + (size_t)t * Dd;
    for (int k = tid; k < Dd; k += 256) {
        float xv = xr[k];
        const float* g = gate + (size_t)k * Ee;
#pragma unroll
        for (int e = 0; e < Ee; ++e) acc[e] = fmaf(xv, g[e], acc[e]);
    }
#pragma unroll
    for (int e = 0; e < Ee; ++e) red[e * 256 + tid] = acc[e];
    __syncthreads();
    for (int s = 128; s > 0; s >>= 1) {
        if (tid < s) {
#pragma unroll
            for (int e = 0; e < Ee; ++e) red[e * 256 + tid] += red[e * 256 + tid + s];
        }
        __syncthreads();
    }
    if (tid == 0) {
        float l[Ee]; float mx = -1e30f;
#pragma unroll
        for (int e = 0; e < Ee; ++e) { l[e] = red[e * 256]; mx = fmaxf(mx, l[e]); }
        float Z = 0.f;
#pragma unroll
        for (int e = 0; e < Ee; ++e) { l[e] = expf(l[e] - mx); Z += l[e]; }
#pragma unroll
        for (int e = 0; e < Ee; ++e) l[e] /= Z;
        int i0 = 0;
#pragma unroll
        for (int e = 1; e < Ee; ++e) if (l[e] > l[i0]) i0 = e;   // ties -> lowest idx
        int i1 = -1;
#pragma unroll
        for (int e = 0; e < Ee; ++e) {
            if (e == i0) continue;
            if (i1 < 0 || l[e] > l[i1]) i1 = e;
        }
        float s0 = l[i0], s1 = l[i1];
        float denom = fmaxf(s0 + s1, 1e-12f);
        top2_e[t * 2 + 0] = i0;  top2_e[t * 2 + 1] = i1;
        top2_w[t * 2 + 0] = s0 / denom;  top2_w[t * 2 + 1] = s1 / denom;
        atomicAdd(&cnt[i0], 1);
        atomicAdd(&cnt[i1], 1);
    }
}

// ---------------- 2) single-thread tile plan (E=8, trivial) ----------------
__global__ void plan_kernel(const int* __restrict__ cnt,
                            int* __restrict__ seg_off, int* __restrict__ cursor,
                            int* __restrict__ tile_e, int* __restrict__ tile_row,
                            int* __restrict__ ntiles, int* __restrict__ row_token) {
    if (threadIdx.x != 0 || blockIdx.x != 0) return;
    int run = 0, tt = 0;
    for (int e = 0; e < Ee; ++e) {
        seg_off[e] = run;
        cursor[e]  = 0;
        int c  = cnt[e];
        int pc = ((c + BM - 1) / BM) * BM;          // pad segment to BM
        for (int j = 0; j < pc / BM; ++j) { tile_e[tt] = e; tile_row[tt] = run + j * BM; ++tt; }
        for (int r = c; r < pc; ++r) row_token[run + r] = -1;   // padding rows
        run += pc;
    }
    ntiles[0] = tt;
}

// ---------------- 3) scatter tokens into per-expert row slots ----------------
__global__ void scatter_kernel(const int* __restrict__ top2_e,
                               int* __restrict__ cursor, const int* __restrict__ seg_off,
                               int* __restrict__ row_token, int* __restrict__ tok_pos) {
    int t = blockIdx.x * blockDim.x + threadIdx.x;
    if (t >= Nn) return;
#pragma unroll
    for (int k = 0; k < 2; ++k) {
        int e   = top2_e[t * 2 + k];
        int pos = seg_off[e] + atomicAdd(&cursor[e], 1);
        row_token[pos]     = t;
        tok_pos[t * 2 + k] = pos;
    }
}

// ---------------- 4a) pack x fp32 -> bf16 (row-major), once ----------------
__global__ void pack_x_kernel(const float* __restrict__ x, bf16* __restrict__ Xb) {
    int i = blockIdx.x * blockDim.x + threadIdx.x;   // one 8-elem chunk
    const float* s = x + (size_t)i * 8;
    v8bf o;
#pragma unroll
    for (int j = 0; j < 8; ++j) o[j] = f2bf(s[j]);
    *(v8bf*)(Xb + (size_t)i * 8) = o;
}

// ---------------- 4b) pack weights fp32 -> bf16 in WMMA B-fragment order ------
// Packed block (e, n-tile of 16, k-tile of 32): 32 lanes x 16 bf16 contiguous.
// Lane L = (half = L/16, n = L%16): values j=0..15 -> K = kt*32 + 16*half + j.
__global__ void pack_w_kernel(const float* __restrict__ W, bf16* __restrict__ Wp,
                              int Kdim, int Ndim) {
    const int Kt = Kdim / 32, Nt = Ndim / 16;
    int chunk = blockIdx.x * blockDim.x + threadIdx.x;     // one lane-chunk
    int total = Ee * Nt * Kt * 32;
    if (chunk >= total) return;
    int lane = chunk & 31;
    int kt   = (chunk >> 5) % Kt;
    int nt   = (chunk / (32 * Kt)) % Nt;
    int e    =  chunk / (32 * Kt * Nt);
    int n    = nt * 16 + (lane & 15);
    int half = lane >> 4;
    const float* src = W + (size_t)e * Kdim * Ndim;
    bf16* dst = Wp + (size_t)chunk * 16;
    v8bf o0, o1;
#pragma unroll
    for (int j = 0; j < 8; ++j)
        o0[j] = f2bf(src[(size_t)(kt * 32 + 16 * half + j) * Ndim + n]);
#pragma unroll
    for (int j = 0; j < 8; ++j)
        o1[j] = f2bf(src[(size_t)(kt * 32 + 16 * half + 8 + j) * Ndim + n]);
    *(v8bf*)(dst)     = o0;
    *(v8bf*)(dst + 8) = o1;
}

// ---- 5) grouped WMMA GEMM, async double-buffered A tiles ----
// stage 1: h = relu(x[tok] @ W1[e] + b1[e])  (A gathered via row_token, bf16 x)
// stage 2: y = h @ W2[e] + b2[e]             (A = Hbuf rows, identity)
template <int STAGE>
__global__ void moe_gemm_kernel(const bf16*  __restrict__ Axb,   // stage1 A source (Xb)
                                const bf16*  __restrict__ Hin,   // stage2 A source
                                const bf16*  __restrict__ zpage, // 256B of zeros
                                const bf16*  __restrict__ Wp,
                                const float* __restrict__ bias,
                                bf16*  __restrict__ Hout,
                                float* __restrict__ Ybuf,
                                const int* __restrict__ row_token,
                                const int* __restrict__ tile_e,
                                const int* __restrict__ tile_row,
                                const int* __restrict__ ntiles) {
    constexpr int Kdim = (STAGE == 1) ? Dd : Hh;
    constexpr int Ndim = (STAGE == 1) ? Hh : Dd;
    constexpr int Ktiles   = Kdim / 32;
    constexpr int Ntiles16 = Ndim / 16;

    if ((int)blockIdx.x >= ntiles[0]) return;
    const int e    = tile_e[blockIdx.x];
    const int row0 = tile_row[blockIdx.x];
    const int n0   = blockIdx.y * BN;

    __shared__ bf16 As[2][BM][BK];       // 2 x 8 KB, double buffered

    const int tid  = threadIdx.x;        // 0..255 (8 waves, wave32)
    const int lane = tid & 31;
    const int wave = tid >> 5;
    const int wm   = wave >> 1;          // 0..3 -> 32 rows each
    const int wn   = wave & 1;           // 0..1 -> 64 cols each
    const int half = lane >> 4;
    const int l16  = lane & 15;

    // ---- per-thread fixed copy slot: half-row (32 bytes) of the A tile ----
    const int cm = tid >> 1;             // row within tile
    const int hf = tid & 1;              // which 16-elem half
    unsigned long long abase;            // global byte address of this row
    unsigned kmul;                       // bytes advanced per k-step (0 for pad rows)
    if (STAGE == 1) {
        const int tok = row_token[row0 + cm];
        if (tok >= 0) { abase = (unsigned long long)(uintptr_t)(Axb + (size_t)tok * Kdim); kmul = BK * 2; }
        else          { abase = (unsigned long long)(uintptr_t)zpage;                      kmul = 0; }
    } else {
        abase = (unsigned long long)(uintptr_t)(Hin + (size_t)(row0 + cm) * Kdim);
        kmul  = BK * 2;
    }
    const unsigned lds0 = lds_offset_of(&As[0][cm][hf * 16]);
    const unsigned lds1 = lds_offset_of(&As[1][cm][hf * 16]);

    v8f acc[2][4];
#pragma unroll
    for (int mt = 0; mt < 2; ++mt)
#pragma unroll
        for (int nt = 0; nt < 4; ++nt)
#pragma unroll
            for (int v = 0; v < 8; ++v) acc[mt][nt][v] = 0.f;

    // ---- prologue: stream tile 0 into buffer 0 ----
    {
        unsigned long long ga = abase + (unsigned long long)(hf * 32);
        async_cp16(lds0, ga);
        async_cp16_o16(lds0, ga);
    }
    wait_async0();
    __syncthreads();

    for (int kt = 0; kt < Ktiles; ++kt) {
        const int cur = kt & 1;
        // ---- async prefetch of next A tile into the other buffer ----
        if (kt + 1 < Ktiles) {
            unsigned long long ga = abase + (unsigned long long)kmul * (kt + 1)
                                          + (unsigned long long)(hf * 32);
            unsigned ldsn = cur ? lds0 : lds1;
            async_cp16(ldsn, ga);
            async_cp16_o16(ldsn, ga);
        }

        // ---- A fragments from LDS (contiguous k = 16*half + j, matching packed B) ----
        v16bf afrag[2];
#pragma unroll
        for (int mt = 0; mt < 2; ++mt) {
            const int mrow = wm * 32 + mt * 16 + l16;
            v8bf lo = *(const v8bf*)&As[cur][mrow][16 * half];
            v8bf hi = *(const v8bf*)&As[cur][mrow][16 * half + 8];
            afrag[mt] = __builtin_shufflevector(lo, hi,
                0, 1, 2, 3, 4, 5, 6, 7, 8, 9, 10, 11, 12, 13, 14, 15);
        }
        // ---- B fragments straight from L2-resident packed weights ----
#pragma unroll
        for (int nt = 0; nt < 4; ++nt) {
            const int ntg = (n0 >> 4) + wn * 4 + nt;
            const bf16* bp = Wp + ((size_t)((e * Ntiles16 + ntg) * Ktiles + kt)) * 512
                                + lane * 16;
            if (kt + 1 < Ktiles) __builtin_prefetch((const char*)bp + 1024, 0, 0);
            v8bf lo = *(const v8bf*)(bp);
            v8bf hi = *(const v8bf*)(bp + 8);
            v16bf bfrag = __builtin_shufflevector(lo, hi,
                0, 1, 2, 3, 4, 5, 6, 7, 8, 9, 10, 11, 12, 13, 14, 15);
#pragma unroll
            for (int mt = 0; mt < 2; ++mt)
                acc[mt][nt] = __builtin_amdgcn_wmma_f32_16x16x32_bf16(
                    false, afrag[mt], false, bfrag, (short)0, acc[mt][nt], false, false);
        }

        // ---- close the pipeline stage: async writes landed, all waves synced ----
        wait_async0();
        __syncthreads();
    }

    // ---- epilogue: bias (+ReLU/convert) using C layout: n=lane%16, m=v+8*half --
#pragma unroll
    for (int nt = 0; nt < 4; ++nt) {
        const int gn = n0 + wn * 64 + nt * 16 + l16;
        const float bv = bias[e * Ndim + gn];
#pragma unroll
        for (int mt = 0; mt < 2; ++mt) {
#pragma unroll
            for (int v = 0; v < 8; ++v) {
                const int gm = row0 + wm * 32 + mt * 16 + v + 8 * half;
                float val = acc[mt][nt][v] + bv;
                if (STAGE == 1) {
                    val = fmaxf(val, 0.f);
                    Hout[(size_t)gm * Hh + gn] = f2bf(val);
                } else {
                    Ybuf[(size_t)gm * Dd + gn] = val;
                }
            }
        }
    }
}

// ---------------- 6) weighted top-2 combine ----------------
__global__ void combine_kernel(const float* __restrict__ Ybuf,
                               const int* __restrict__ tok_pos,
                               const float* __restrict__ top2_w,
                               float* __restrict__ out) {
    int idx = blockIdx.x * blockDim.x + threadIdx.x;   // over Nn * Dd/4
    int t  = idx / (Dd / 4);
    int c4 = idx % (Dd / 4);
    if (t >= Nn) return;
    const int   p0 = tok_pos[t * 2 + 0], p1 = tok_pos[t * 2 + 1];
    const float w0 = top2_w[t * 2 + 0], w1 = top2_w[t * 2 + 1];
    v4f y0 = *((const v4f*)(Ybuf + (size_t)p0 * Dd) + c4);
    v4f y1 = *((const v4f*)(Ybuf + (size_t)p1 * Dd) + c4);
    v4f r  = y0 * w0 + y1 * w1;
    *((v4f*)(out + (size_t)t * Dd) + c4) = r;
}

// ---------------- host-side launcher ----------------
extern "C" void kernel_launch(void* const* d_in, const int* in_sizes, int n_in,
                              void* d_out, int out_size, void* d_ws, size_t ws_size,
                              hipStream_t stream) {
    (void)in_sizes; (void)n_in; (void)out_size; (void)ws_size;
    const float* x    = (const float*)d_in[0];
    const float* gate = (const float*)d_in[1];
    const float* W1   = (const float*)d_in[2];
    const float* b1   = (const float*)d_in[3];
    const float* W2   = (const float*)d_in[4];
    const float* b2   = (const float*)d_in[5];
    float* out = (float*)d_out;

    char* ws = (char*)d_ws;
    size_t off = 0;
    auto alloc = [&](size_t bytes) -> void* {
        void* p = ws + off;
        off = (off + bytes + 255) & ~(size_t)255;
        return p;
    };
    bf16*  W1p       = (bf16*)alloc(sizeof(bf16) * (size_t)Ee * Dd * Hh);   // 32 MB
    bf16*  W2p       = (bf16*)alloc(sizeof(bf16) * (size_t)Ee * Hh * Dd);   // 32 MB
    bf16*  Xb        = (bf16*)alloc(sizeof(bf16) * (size_t)Nn * Dd);        // 16 MB
    bf16*  Hbuf      = (bf16*)alloc(sizeof(bf16) * (size_t)ROWS_CAP * Hh);  // 68 MB
    float* Ybuf      = (float*)alloc(sizeof(float) * (size_t)ROWS_CAP * Dd);// 68 MB
    bf16*  zpage     = (bf16*)alloc(256);                                   // zero page
    int*   top2_e    = (int*)alloc(sizeof(int) * 2 * Nn);
    float* top2_w    = (float*)alloc(sizeof(float) * 2 * Nn);
    int*   tok_pos   = (int*)alloc(sizeof(int) * 2 * Nn);
    int*   row_token = (int*)alloc(sizeof(int) * ROWS_CAP);
    int*   cnt       = (int*)alloc(sizeof(int) * Ee);
    int*   cursor    = (int*)alloc(sizeof(int) * Ee);
    int*   seg_off   = (int*)alloc(sizeof(int) * Ee);
    int*   ntiles    = (int*)alloc(sizeof(int));
    int*   tile_e    = (int*)alloc(sizeof(int) * MAXTILES);
    int*   tile_row  = (int*)alloc(sizeof(int) * MAXTILES);

    hipMemsetAsync(cnt, 0, sizeof(int) * Ee, stream);
    hipMemsetAsync(zpage, 0, 256, stream);

    // one-shot conversions / packing (bf16 weights stay L2-resident: 64 MB < 192 MB)
    {
        const int chunks = Ee * (Hh / 16) * (Dd / 32) * 32;   // 1,048,576
        pack_w_kernel<<<chunks / 256, 256, 0, stream>>>(W1, W1p, Dd, Hh);
    }
    {
        const int chunks = Ee * (Dd / 16) * (Hh / 32) * 32;   // 1,048,576
        pack_w_kernel<<<chunks / 256, 256, 0, stream>>>(W2, W2p, Hh, Dd);
    }
    pack_x_kernel<<<(Nn * Dd / 8) / 256, 256, 0, stream>>>(x, Xb);

    gate_topk_kernel<<<Nn, 256, 0, stream>>>(x, gate, top2_e, top2_w, cnt);
    plan_kernel<<<1, 32, 0, stream>>>(cnt, seg_off, cursor, tile_e, tile_row, ntiles, row_token);
    scatter_kernel<<<(Nn + 255) / 256, 256, 0, stream>>>(top2_e, cursor, seg_off, row_token, tok_pos);

    dim3 g1(MAXTILES, Hh / BN);   // 136 x 16
    moe_gemm_kernel<1><<<g1, 256, 0, stream>>>(Xb, nullptr, zpage, W1p, b1, Hbuf, nullptr,
                                               row_token, tile_e, tile_row, ntiles);
    dim3 g2(MAXTILES, Dd / BN);   // 136 x 8
    moe_gemm_kernel<2><<<g2, 256, 0, stream>>>(nullptr, Hbuf, zpage, W2p, b2, nullptr, Ybuf,
                                               row_token, tile_e, tile_row, ntiles);

    combine_kernel<<<(Nn * (Dd / 4) + 255) / 256, 256, 0, stream>>>(Ybuf, tok_pos, top2_w, out);
}
